// AttentionHead_10926396801438
// MI455X (gfx1250) — matrix operand order repaired and checked by reading, not compiled, not run on hardware
//
#include <hip/hip_runtime.h>

typedef _Float16 half_t;
typedef __attribute__((ext_vector_type(16))) _Float16 v16h;
typedef __attribute__((ext_vector_type(8)))  _Float16 v8h;
typedef __attribute__((ext_vector_type(8)))  float    v8f;
typedef __attribute__((ext_vector_type(4)))  float    v4f;

#define B_ 8
#define T_ 2048
#define C_ 1024
#define H_ 64

__device__ inline v16h join8(v8h lo, v8h hv) {
    v16h r;
    #pragma unroll
    for (int i = 0; i < 8; ++i) { r[i] = lo[i]; r[8 + i] = hv[i]; }
    return r;
}

// ---------------------------------------------------------------------------
// Kernel 1: fused QKV projection.  a[16384,1024] f32 x W[1024,64] f32 (x3)
// -> K row-major f16, Q row-major f16 (pre-scaled by H^-0.5), V transposed
// f16 [B,64,T].  256 blocks x 128 threads; each wave owns a 16-row tile and
// computes 12 output tiles (3 matrices x 4 col-tiles), sharing one A-frag.
// ---------------------------------------------------------------------------
#define K1_KC 128
#define K1_WS 136   // 128 + 8 pad (f16): 68-dword row stride -> conflict-free b128

__global__ __launch_bounds__(128) void qkv_kernel(
    const float* __restrict__ a,  const float* __restrict__ Wk,
    const float* __restrict__ Wq, const float* __restrict__ Wv,
    half_t* __restrict__ kbuf, half_t* __restrict__ qbuf, half_t* __restrict__ vtbuf)
{
    __shared__ half_t ldsW[3 * 64 * K1_WS];
    const int tid  = threadIdx.x;
    const int lane = tid & 31;
    const int wv   = tid >> 5;
    const int rowbase = blockIdx.x * 64 + wv * 16;
    const int m  = lane & 15;
    const int hh = lane >> 4;   // lane half: 0 or 1

    const v8f vzero = {};
    v8f acc[12];
    #pragma unroll
    for (int i = 0; i < 12; ++i) acc[i] = vzero;

    for (int kc = 0; kc < C_; kc += K1_KC) {
        // stage W chunk (3 x 128 x 64 f32) into LDS, transposed [mat][h][k] f16
        for (int i = 0; i < (3 * K1_KC * 64) / 128; ++i) {
            int idx = i * 128 + tid;
            int mat = idx >> 13;          // 128*64 = 8192 per matrix
            int rem = idx & 8191;
            int k   = rem >> 6;
            int h   = rem & 63;
            const float* Wp = (mat == 0) ? Wk : ((mat == 1) ? Wq : Wv);
            ldsW[mat * (64 * K1_WS) + h * K1_WS + k] = (half_t)Wp[(kc + k) * H_ + h];
        }
        __syncthreads();

        // prefetch next chunk of W into cache while this chunk computes
        if (kc + K1_KC < C_) {
            const float* nk = Wk + (size_t)(kc + K1_KC) * H_ + tid * 64;
            const float* nq = Wq + (size_t)(kc + K1_KC) * H_ + tid * 64;
            const float* nv = Wv + (size_t)(kc + K1_KC) * H_ + tid * 64;
            __builtin_prefetch(nk, 0, 1);
            __builtin_prefetch(nq, 0, 1);
            __builtin_prefetch(nv, 0, 1);
        }

        #pragma unroll
        for (int ks = 0; ks < K1_KC / 32; ++ks) {
            // A fragment: 16x32 f16, rows = this wave's 16 rows, from global f32
            const float* arow = a + (size_t)(rowbase + m) * C_ + kc + ks * 32 + hh * 8;
            v4f f0 = *(const v4f*)(arow);
            v4f f1 = *(const v4f*)(arow + 4);
            v4f f2 = *(const v4f*)(arow + 16);
            v4f f3 = *(const v4f*)(arow + 20);
            v16h A;
            #pragma unroll
            for (int i = 0; i < 4; ++i) {
                A[i]      = (half_t)f0[i];
                A[4 + i]  = (half_t)f1[i];
                A[8 + i]  = (half_t)f2[i];
                A[12 + i] = (half_t)f3[i];
            }
            // software-pipelined B fragments: load t+1 while WMMA t executes
            const half_t* base = &ldsW[m * K1_WS + ks * 32 + hh * 8];
            v16h Bv = join8(*(const v8h*)base, *(const v8h*)(base + 16));
            #pragma unroll
            for (int t = 0; t < 12; ++t) {
                v16h Bn = Bv;
                if (t < 11) {
                    int mat = (t + 1) >> 2, co = (t + 1) & 3;
                    const half_t* bp = base + mat * (64 * K1_WS) + co * 16 * K1_WS;
                Bn = join8(*(const v8h*)bp, *(const v8h*)(bp + 16));
                }
                acc[t] = __builtin_amdgcn_wmma_f32_16x16x32_f16(
                    false, A, false, Bv, (short)0, acc[t], false, false);
                Bv = Bn;
            }
        }
        __syncthreads();
    }

    // D layout: vgpr r -> row r (lanes 0-15) / row 8+r (lanes 16-31); col = lane&15
    #pragma unroll
    for (int mat = 0; mat < 3; ++mat) {
        #pragma unroll
        for (int co = 0; co < 4; ++co) {
            v8f t = acc[mat * 4 + co];
            #pragma unroll
            for (int r = 0; r < 8; ++r) {
                int row = rowbase + r + 8 * hh;
                int col = co * 16 + m;
                if (mat == 0)      kbuf[row * H_ + col] = (half_t)t[r];
                else if (mat == 1) qbuf[row * H_ + col] = (half_t)(t[r] * 0.125f);
                else {
                    int bb = row >> 11, tt = row & (T_ - 1);
                    vtbuf[((size_t)(bb * H_ + col)) * T_ + tt] = (half_t)t[r];
                }
            }
        }
    }
}

// ---------------------------------------------------------------------------
// Kernel 2: causal flash attention. 256 blocks x 128 threads; each wave owns
// a 16-query tile; keys processed 32 at a time (4 WMMA for S, 4 for O).
// ---------------------------------------------------------------------------
#define K2_PS 40    // P tile LDS stride (32 + 8 pad halves)

__global__ __launch_bounds__(128) void attn_kernel(
    const half_t* __restrict__ qbuf, const half_t* __restrict__ kbuf,
    const half_t* __restrict__ vtbuf, float* __restrict__ out)
{
    __shared__ half_t ldsP[4 * 16 * K2_PS];
    const int tid  = threadIdx.x;
    const int lane = tid & 31;
    const int wv   = tid >> 5;
    const int b    = blockIdx.x >> 5;     // T/64 = 32 query blocks per batch
    const int qblk = blockIdx.x & 31;
    const int qbase = qblk * 64 + wv * 16;
    const int m  = lane & 15;
    const int hh = lane >> 4;
    half_t* pl = &ldsP[wv * 16 * K2_PS];

    // Q fragments (A-type), pre-scaled by H^-0.5 in kernel 1
    const half_t* qrow = qbuf + ((size_t)(b * T_ + qbase + m)) * H_;
    v16h Q[2];
    #pragma unroll
    for (int ks = 0; ks < 2; ++ks) {
        const half_t* p = qrow + ks * 32 + hh * 16;
        Q[ks] = join8(*(const v8h*)p, *(const v8h*)(p + 8));
    }

    const v8f vzero = {};
    v8f O[4];
    #pragma unroll
    for (int i = 0; i < 4; ++i) O[i] = vzero;
    float mrow[8], lrow[8];
    #pragma unroll
    for (int r = 0; r < 8; ++r) { mrow[r] = -1e30f; lrow[r] = 0.0f; }

    const half_t* kb_p = kbuf  + (size_t)b * T_ * H_;
    const half_t* vb_p = vtbuf + (size_t)b * H_ * T_;

    for (int kb = 0; kb < qbase + 16; kb += 32) {
        // ---- preload all 4 K fragments (one clause), then S = Q K^T ----
        v16h Kf[4];
        #pragma unroll
        for (int s = 0; s < 2; ++s) {
            int key = kb + s * 16 + m;            // B-frag: lane -> key column
            #pragma unroll
            for (int ks = 0; ks < 2; ++ks) {
                const half_t* p = kb_p + (size_t)key * H_ + ks * 32 + hh * 16;
                Kf[s * 2 + ks] = join8(*(const v8h*)p, *(const v8h*)(p + 8));
            }
        }
        v8f S[2];
        #pragma unroll
        for (int s = 0; s < 2; ++s) {
            S[s] = vzero;
            #pragma unroll
            for (int ks = 0; ks < 2; ++ks)
                S[s] = __builtin_amdgcn_wmma_f32_16x16x32_f16(
                    false, Q[ks], false, Kf[s * 2 + ks], (short)0, S[s], false, false);
        }
        // ---- issue V fragment loads now; latency hides under softmax ----
        v16h Vf[4];
        #pragma unroll
        for (int ct = 0; ct < 4; ++ct) {
            const half_t* p = vb_p + (size_t)(ct * 16 + m) * T_ + kb + hh * 16;
            Vf[ct] = join8(*(const v8h*)p, *(const v8h*)(p + 8));
        }
        // ---- causal mask (only on diagonal-overlapping tiles; uniform branch) ----
        if (kb + 31 > qbase) {
            #pragma unroll
            for (int s = 0; s < 2; ++s) {
                int key = kb + s * 16 + m;
                #pragma unroll
                for (int r = 0; r < 8; ++r) {
                    int q = qbase + r + 8 * hh;
                    S[s][r] = (key > q) ? -1e30f : S[s][r];
                }
            }
        }
        // ---- online softmax (row stats live per 16-lane half) ----
        float alpha[8];
        #pragma unroll
        for (int r = 0; r < 8; ++r) {
            float t = fmaxf(S[0][r], S[1][r]);
            #pragma unroll
            for (int off = 1; off < 16; off <<= 1) t = fmaxf(t, __shfl_xor(t, off, 32));
            float mn = fmaxf(mrow[r], t);
            alpha[r] = __expf(mrow[r] - mn);
            mrow[r]  = mn;
            float p0 = __expf(S[0][r] - mn);
            float p1 = __expf(S[1][r] - mn);
            S[0][r] = p0; S[1][r] = p1;
            float sum = p0 + p1;
            #pragma unroll
            for (int off = 1; off < 16; off <<= 1) sum += __shfl_xor(sum, off, 32);
            lrow[r] = lrow[r] * alpha[r] + sum;
        }
        // ---- re-layout P (D-layout f32) -> A-layout f16 via per-wave LDS ----
        #pragma unroll
        for (int r = 0; r < 8; ++r) {
            int prow = r + 8 * hh;
            pl[prow * K2_PS + m]      = (half_t)S[0][r];
            pl[prow * K2_PS + 16 + m] = (half_t)S[1][r];
        }
        asm volatile("s_wait_dscnt 0" ::: "memory");
        v16h Pf;
        {
            const half_t* p = pl + m * K2_PS + hh * 16;
            Pf = join8(*(const v8h*)p, *(const v8h*)(p + 8));
        }
        // ---- O = O*alpha + P V ----
        #pragma unroll
        for (int ct = 0; ct < 4; ++ct) {
            #pragma unroll
            for (int r = 0; r < 8; ++r) O[ct][r] *= alpha[r];
            O[ct] = __builtin_amdgcn_wmma_f32_16x16x32_f16(
                false, Pf, false, Vf[ct], (short)0, O[ct], false, false);
        }
    }

    // ---- epilogue: O / l -> out (f32) ----
    float inv[8];
    #pragma unroll
    for (int r = 0; r < 8; ++r) inv[r] = 1.0f / lrow[r];
    #pragma unroll
    for (int ct = 0; ct < 4; ++ct) {
        #pragma unroll
        for (int r = 0; r < 8; ++r) {
            int row = qbase + r + 8 * hh;
            out[((size_t)(b * T_ + row)) * H_ + ct * 16 + m] = O[ct][r] * inv[r];
        }
    }
}

// ---------------------------------------------------------------------------
extern "C" void kernel_launch(void* const* d_in, const int* in_sizes, int n_in,
                              void* d_out, int out_size, void* d_ws, size_t ws_size,
                              hipStream_t stream) {
    (void)in_sizes; (void)n_in; (void)out_size; (void)ws_size;
    const float* a  = (const float*)d_in[0];
    const float* Wk = (const float*)d_in[1];
    const float* Wq = (const float*)d_in[2];
    const float* Wv = (const float*)d_in[3];
    // workspace: K (2MB) | Q (2MB) | Vt (2MB), all f16
    half_t* kb = (half_t*)d_ws;
    half_t* qb = (half_t*)((char*)d_ws + 2097152);
    half_t* vt = (half_t*)((char*)d_ws + 4194304);
    float*  out = (float*)d_out;

    qkv_kernel<<<dim3(256), dim3(128), 0, stream>>>(a, Wk, Wq, Wv, kb, qb, vt);
    attn_kernel<<<dim3(256), dim3(128), 0, stream>>>(qb, kb, vt, out);
}
